// cnnToLSTMCustom_8366596292951
// MI455X (gfx1250) — matrix operand order repaired and checked by reading, not compiled, not run on hardware
//
#include <hip/hip_runtime.h>

typedef __attribute__((ext_vector_type(16))) _Float16 v16h;
typedef __attribute__((ext_vector_type(8)))  _Float16 v8h;
typedef __attribute__((ext_vector_type(16))) short    v16s;
typedef __attribute__((ext_vector_type(8)))  float    v8f;

#define NB 4       // batch
#define LL 256     // seq len
#define DN 300     // embed / conv channels
#define DNP 320    // padded channels (mult of 32)
#define DOP 304    // padded conv output channels (mult of 16)
#define HH 256     // LSTM hidden
#define GG 1024    // 4*HH gates
#define TT 512     // scatter target length
#define NC_ 128    // PCA components
#define MR 2048    // TT*NB rows of LSTM input
#define NEV 512    // eig problem dim (2*HH)
#define KCOV 32    // padded K for covariance GEMM

// ---------------- WMMA helpers (CDNA5 16x16x32 f16 -> f32) ----------------
__device__ __forceinline__ v8f wmma_acc(v16h a, v16h b, v8f c) {
  return __builtin_amdgcn_wmma_f32_16x16x32_f16(false, a, false, b, (short)0, c,
                                                false, false);
}

// A fragment 16x32 (MxK), row-major, REQUIRES: 16 valid rows at m0, lda%8==0,
// 32 valid (possibly zero-padded) halves at k0. Two contiguous 16B loads/lane.
__device__ __forceinline__ v16h fragA(const _Float16* __restrict__ A, int lda,
                                      int m0, int k0, int lane) {
  const _Float16* p = A + (size_t)(m0 + (lane & 15)) * lda + (k0 + ((lane >> 4) << 4));
  v8h lo = *(const v8h*)p;
  v8h hi = *(const v8h*)(p + 8);
  return __builtin_shufflevector(lo, hi, 0, 1, 2, 3, 4, 5, 6, 7,
                                 8, 9, 10, 11, 12, 13, 14, 15);
}

// B fragment 32x16 (KxN) sourced from B^T (N-major, ldbt%8==0): 16 contiguous
// halves per lane (two 16B loads).
__device__ __forceinline__ v16h fragBT(const _Float16* __restrict__ BT, int ldbt,
                                       int n0, int k0, int lane) {
  const _Float16* p = BT + (size_t)(n0 + (lane & 15)) * ldbt + (k0 + ((lane >> 4) << 4));
  v8h lo = *(const v8h*)p;
  v8h hi = *(const v8h*)(p + 8);
  return __builtin_shufflevector(lo, hi, 0, 1, 2, 3, 4, 5, 6, 7,
                                 8, 9, 10, 11, 12, 13, 14, 15);
}

__device__ __forceinline__ v16h negfrag(v16h f) {   // flip fp16 sign bits
  v16s u = __builtin_bit_cast(v16s, f);
  u ^= (short)0x8000;
  return __builtin_bit_cast(v16h, u);
}

// ---------------- complex scalar helpers ----------------
struct cpx { float re, im; };
__device__ __forceinline__ cpx cmulc(cpx a, cpx b) {
  return {a.re * b.re - a.im * b.im, a.re * b.im + a.im * b.re};
}
__device__ __forceinline__ cpx cinvc(cpx z) {
  float d = z.re * z.re + z.im * z.im;
  d = (d > 1e-30f) ? d : 1e-30f;
  return {z.re / d, -z.im / d};
}
__device__ __forceinline__ cpx csigc(cpx z) {       // 1/(1+exp(-z))
  float ex = __expf(-z.re);
  float sn = sinf(-z.im), cs = cosf(-z.im);
  cpx den = {1.f + ex * cs, ex * sn};
  return cinvc(den);
}
__device__ __forceinline__ cpx ctanhc(cpx z) {      // (tanh x + i tan y)/(1 + i tanh x tan y)
  float tx = tanhf(z.re);
  float ty = tanf(z.im);
  cpx num = {tx, ty};
  cpx den = {1.f, tx * ty};
  return cmulc(num, cinvc(den));
}

// ---------------- optional TDM probe (never launched; emits tensor ops) --------------
#if defined(__has_builtin)
#if __has_builtin(__builtin_amdgcn_tensor_load_to_lds) && \
    __has_builtin(__builtin_amdgcn_s_wait_tensorcnt)
#define HAVE_TDM 1
#endif
#endif
#ifdef HAVE_TDM
typedef __attribute__((ext_vector_type(4))) unsigned int u32x4;
typedef __attribute__((ext_vector_type(8))) int          i32x8;
typedef __attribute__((ext_vector_type(4))) int          i32x4;
__global__ void k_tdm_probe(const unsigned int* __restrict__ desc) {
  // Builds a D# from memory and issues one tensor DMA; kernel is never launched.
  u32x4 g0 = {desc[0], desc[1], desc[2], desc[3]};
  i32x8 g1 = {(int)desc[4], (int)desc[5], (int)desc[6], (int)desc[7],
              (int)desc[8], (int)desc[9], (int)desc[10], (int)desc[11]};
  i32x4 g2 = {(int)desc[12], (int)desc[13], (int)desc[14], (int)desc[15]};
  i32x4 g3 = {(int)desc[16], (int)desc[17], (int)desc[18], (int)desc[19]};
  i32x8 g4 = {(int)desc[20], (int)desc[21], (int)desc[22], (int)desc[23],
              (int)desc[24], (int)desc[25], (int)desc[26], (int)desc[27]};
  __builtin_amdgcn_tensor_load_to_lds(g0, g1, g2, g3, g4, 0);
  __builtin_amdgcn_s_wait_tensorcnt(0);
}
#endif

// ---------------- staging kernels (all outputs zero-padded) ----------------
// e_nat: [NB][LL][DNP] f16, natural order (d contiguous) for conv B^T loads
__global__ void k_embed16(const int* __restrict__ x, const float* __restrict__ tab,
                          _Float16* __restrict__ e) {
  int idx = blockIdx.x * blockDim.x + threadIdx.x;
  if (idx >= NB * LL * DNP) return;
  int d = idx % DNP;
  int l = (idx / DNP) % LL;
  int n = idx / (DNP * LL);
  float v = 0.f;
  if (d < DN) v = tab[(size_t)x[n * LL + l] * DN + d];
  e[idx] = (_Float16)v;
}

__global__ void k_zero_h(_Float16* p, int n) {
  int i = blockIdx.x * blockDim.x + threadIdx.x;
  if (i < n) p[i] = (_Float16)0.f;
}

// conv weights: [dout][din][taps] f32 -> [tap][DOP][DNP] f16 zero-padded
__global__ void k_convw16(const float* __restrict__ w, _Float16* __restrict__ o, int taps) {
  int idx = blockIdx.x * blockDim.x + threadIdx.x;
  if (idx >= taps * DOP * DNP) return;
  int din = idx % DNP;
  int dout = (idx / DNP) % DOP;
  int tap = idx / (DNP * DOP);
  float v = 0.f;
  if (dout < DN && din < DN) v = w[((size_t)dout * DN + din) * taps + tap];
  o[idx] = (_Float16)v;
}

// Wih [6][GG][DN][2] f32 -> B^T planes [6][GG][DNP] f16 (native N-major, pad K)
__global__ void k_wihBT(const float* __restrict__ W, _Float16* __restrict__ ore,
                        _Float16* __restrict__ oim) {
  int idx = blockIdx.x * blockDim.x + threadIdx.x;
  if (idx >= 6 * GG * DNP) return;
  int k = idx % DNP;
  int g = (idx / DNP) % GG;
  int slab = idx / (DNP * GG);
  float vr = 0.f, vi = 0.f;
  if (k < DN) {
    size_t s = ((size_t)(slab * GG + g) * DN + k) * 2;
    vr = W[s]; vi = W[s + 1];
  }
  ore[idx] = (_Float16)vr;
  oim[idx] = (_Float16)vi;
}

// Whh [6][GG][HH][2] f32 -> B^T planes [6][GG][HH] f16 (pure cast, layout native)
__global__ void k_whhBT(const float* __restrict__ W, _Float16* __restrict__ ore,
                        _Float16* __restrict__ oim) {
  int idx = blockIdx.x * blockDim.x + threadIdx.x;
  if (idx >= 6 * GG * HH) return;
  ore[idx] = (_Float16)W[2 * idx];
  oim[idx] = (_Float16)W[2 * idx + 1];
}

__global__ void k_bcomb(const float* __restrict__ bih, const float* __restrict__ bhh,
                        float* __restrict__ bre, float* __restrict__ bim) {
  int idx = blockIdx.x * blockDim.x + threadIdx.x;
  if (idx >= 6 * GG) return;
  bre[idx] = bih[2 * idx] + bhh[2 * idx];
  bim[idx] = bih[2 * idx + 1] + bhh[2 * idx + 1];
}

// ------------- generic WMMA GEMM: C[MxN] f32 = A[MxK] * B[KxN], B given as B^T -------------
// NEG/ACC are compile-time: no cndmask selection, dead code eliminated.
template <int NEG, int ACC>
__global__ __launch_bounds__(32)
void k_gemm16(const _Float16* __restrict__ A, int lda,
              const _Float16* __restrict__ BT, int ldbt,
              float* __restrict__ C, int ldc,
              int M, int Nn, int K) {
  int lane = threadIdx.x;
  int n0 = blockIdx.x * 16;
  int m0 = blockIdx.y * 16;
  v8f acc = {};
  for (int k0 = 0; k0 < K; k0 += 32) {
    if (k0 + 32 < K) {
      __builtin_prefetch(&A[(size_t)m0 * lda + k0 + 32], 0, 1);
      __builtin_prefetch(&BT[(size_t)n0 * ldbt + k0 + 32], 0, 1);
    }
    v16h a = fragA(A, lda, m0, k0, lane);
    if constexpr (NEG) a = negfrag(a);
    v16h b = fragBT(BT, ldbt, n0, k0, lane);
    acc = wmma_acc(a, b, acc);
  }
  int n = n0 + (lane & 15);
  if (n < Nn) {
#pragma unroll
    for (int v = 0; v < 8; ++v) {
      int m = m0 + (lane >> 4) * 8 + v;
      if (m < M) {
        float r = acc[v];
        if constexpr (ACC) r += C[(size_t)m * ldc + n];
        C[(size_t)m * ldc + n] = r;
      }
    }
  }
}

// ---------------- conv1d as per-tap WMMA GEMM ----------------
// out[n][dout][t] = sum_tap sum_din w16[tap][dout][din] * e[n][stride*t+tap-pad][din] + b[dout]
__global__ __launch_bounds__(32)
void k_conv_wmma(const _Float16* __restrict__ e, const _Float16* __restrict__ w16,
                 const float* __restrict__ bias, float* __restrict__ out,
                 int Lout, int taps, int stride, int pad) {
  int lane = threadIdx.x;
  int t0 = blockIdx.x * 16;
  int m0 = blockIdx.y * 16;
  int nb = blockIdx.z;
  v8f acc = {};
  int t = t0 + (lane & 15);
  int kg = (lane >> 4) << 4;
  for (int tap = 0; tap < taps; ++tap) {
    const _Float16* Wk = w16 + (size_t)tap * DOP * DNP;
    int l = stride * t + tap - pad;
    bool ok = (t < Lout) && (l >= 0) && (l < LL);
    const _Float16* prow = e + ((size_t)nb * LL + (ok ? l : 0)) * DNP + kg;
    v8h zz = {};
    for (int k0 = 0; k0 < DNP; k0 += 32) {
      v16h a = fragA(Wk, DNP, m0, k0, lane);
      v8h lo = *(const v8h*)(prow + k0);
      v8h hi = *(const v8h*)(prow + k0 + 8);
      lo = ok ? lo : zz;
      hi = ok ? hi : zz;
      v16h b = __builtin_shufflevector(lo, hi, 0, 1, 2, 3, 4, 5, 6, 7,
                                       8, 9, 10, 11, 12, 13, 14, 15);
      acc = wmma_acc(a, b, acc);
    }
  }
  if (t < Lout) {
#pragma unroll
    for (int v = 0; v < 8; ++v) {
      int m = m0 + (lane >> 4) * 8 + v;
      if (m < DN) out[((size_t)nb * DN + m) * Lout + t] = acc[v] + bias[m];
    }
  }
}

// ---------------- scatter kernels (seq rows = t*NB+n, padded cols DNP) ----------------
__global__ void k_scat2(const float* __restrict__ y, _Float16* __restrict__ sre) {
  const int NI = 255;
  int idx = blockIdx.x * blockDim.x + threadIdx.x;
  if (idx >= NB * DN * NI) return;
  int i = idx % NI; int d = (idx / NI) % DN; int n = idx / (NI * DN);
  _Float16 v = (_Float16)y[((size_t)n * DN + d) * NI + i];
  sre[(size_t)((2 * i + 1) * NB + n) * DNP + d] = v;
  sre[(size_t)((2 * i + 2) * NB + n) * DNP + d] = v;
}
__global__ void k_scat4(const float* __restrict__ y, _Float16* __restrict__ sim) {
  const int NI = 127;
  int idx = blockIdx.x * blockDim.x + threadIdx.x;
  if (idx >= NB * DN * NI) return;
  int i = idx % NI; int d = (idx / NI) % DN; int n = idx / (NI * DN);
  _Float16 v = (_Float16)y[((size_t)n * DN + d) * NI + i];
  const int offs[4] = {1, 3, 4, 6};
#pragma unroll
  for (int j = 0; j < 4; ++j)
    sim[(size_t)((4 * i + offs[j]) * NB + n) * DNP + d] = v;
}
__global__ void k_scat3(const float* __restrict__ y, _Float16* __restrict__ sre) {
  const int LO = 86, NI = 84;
  int idx = blockIdx.x * blockDim.x + threadIdx.x;
  if (idx >= NB * DN * NI) return;
  int i = 1 + (idx % NI); int d = (idx / NI) % DN; int n = idx / (NI * DN);
  _Float16 v = (_Float16)y[((size_t)n * DN + d) * LO + i];
  const int offs[3] = {-3, -1, 1};
#pragma unroll
  for (int j = 0; j < 3; ++j)
    sre[(size_t)((6 * i + offs[j]) * NB + n) * DNP + d] = v;
}
__global__ void k_scat6(const float* __restrict__ y, _Float16* __restrict__ sim) {
  const int LO = 85, NI = 85;
  int idx = blockIdx.x * blockDim.x + threadIdx.x;
  if (idx >= NB * DN * NI) return;
  int i = idx % NI; int d = (idx / NI) % DN; int n = idx / (NI * DN);
  _Float16 v = (_Float16)y[((size_t)n * DN + d) * LO + i];
  if (i == 0) {
    const int c0[4] = {1, 2, 4, 6};
#pragma unroll
    for (int j = 0; j < 4; ++j)
      sim[(size_t)(c0[j] * NB + n) * DNP + d] = v;
  } else if (i <= 83) {
    const int offs[6] = {-3, -1, 1, 2, 4, 6};
#pragma unroll
    for (int j = 0; j < 6; ++j)
      sim[(size_t)((6 * i + offs[j]) * NB + n) * DNP + d] = v;
  } else {
    const int cl[4] = {TT - 7, TT - 5, TT - 3, TT - 2};
#pragma unroll
    for (int j = 0; j < 4; ++j)
      sim[(size_t)(cl[j] * NB + n) * DNP + d] = v;
  }
}
__global__ void k_scat5(const float* __restrict__ y, _Float16* __restrict__ sim) {
  const int LO = 84, NI = 83;
  int idx = blockIdx.x * blockDim.x + threadIdx.x;
  if (idx >= NB * DN * NI) return;
  int i = 1 + (idx % NI); int d = (idx / NI) % DN; int n = idx / (NI * DN);
  _Float16 v = (_Float16)y[((size_t)n * DN + d) * LO + i];
  const int offs[5] = {1, 3, 5, 6, 8};
#pragma unroll
  for (int j = 0; j < 5; ++j)
    sim[(size_t)((6 * (i - 1) + offs[j]) * NB + n) * DNP + d] = v;
}

__global__ void k_addbias(float* __restrict__ pre_re, float* __restrict__ pre_im,
                          const float* __restrict__ bre, const float* __restrict__ bim) {
  int idx = blockIdx.x * blockDim.x + threadIdx.x;
  if (idx >= MR * GG) return;
  int g = idx % GG;
  pre_re[idx] += bre[g];
  pre_im[idx] += bim[g];
}

// ---------------- biLSTM recurrent scan (one workgroup, WMMA per step) ----------------
__global__ __launch_bounds__(512)
void k_lstm_scan(const float* __restrict__ pre_re, const float* __restrict__ pre_im,
                 const _Float16* __restrict__ UBTre, const _Float16* __restrict__ UBTim,
                 float* __restrict__ feat_re, float* __restrict__ feat_im,
                 int reverse, int coloff) {
  __shared__ __align__(16) _Float16 hre[16 * HH];
  __shared__ __align__(16) _Float16 him[16 * HH];
  __shared__ float gre[4 * GG];
  __shared__ float gim[4 * GG];
  int tid = threadIdx.x;
  int lane = tid & 31, wv = tid >> 5;      // 16 waves
  for (int i = tid; i < 16 * HH; i += 512) {
    hre[i] = (_Float16)0.f;
    him[i] = (_Float16)0.f;
  }
  float c_re[4] = {0.f, 0.f, 0.f, 0.f}, c_im[4] = {0.f, 0.f, 0.f, 0.f};
  float s_re[4] = {0.f, 0.f, 0.f, 0.f}, s_im[4] = {0.f, 0.f, 0.f, 0.f};
  __syncthreads();
  for (int step = 0; step < TT; ++step) {
    int t = reverse ? (TT - 1 - step) : step;
    // gates = h @ U^T (complex); 64 col tiles, 4 per wave
#pragma unroll
    for (int ti = 0; ti < 4; ++ti) {
      int n0 = (wv * 4 + ti) * 16;
      v8f ar = {};
      v8f ai = {};
      for (int k0 = 0; k0 < HH; k0 += 32) {
        v16h a_r = fragA(hre, HH, 0, k0, lane);
        v16h a_i = fragA(him, HH, 0, k0, lane);
        v16h a_in = negfrag(a_i);
        v16h b_r = fragBT(UBTre, HH, n0, k0, lane);
        v16h b_i = fragBT(UBTim, HH, n0, k0, lane);
        ar = wmma_acc(a_r, b_r, ar);
        ar = wmma_acc(a_in, b_i, ar);
        ai = wmma_acc(a_r, b_i, ai);
        ai = wmma_acc(a_i, b_r, ai);
      }
      if (lane < 16) {
        int col = n0 + lane;
#pragma unroll
        for (int v = 0; v < 4; ++v) {   // batch rows 0..3 live in lanes 0-15, VGPR v
          gre[v * GG + col] = ar[v] + pre_re[(size_t)(t * NB + v) * GG + col];
          gim[v * GG + col] = ai[v] + pre_im[(size_t)(t * NB + v) * GG + col];
        }
      }
    }
    __syncthreads();
    if (tid < HH) {
      int u = tid;
#pragma unroll
      for (int r = 0; r < 4; ++r) {
        cpx gi = {gre[r * GG + u], gim[r * GG + u]};
        cpx gf = {gre[r * GG + HH + u], gim[r * GG + HH + u]};
        cpx gc = {gre[r * GG + 2 * HH + u], gim[r * GG + 2 * HH + u]};
        cpx go = {gre[r * GG + 3 * HH + u], gim[r * GG + 3 * HH + u]};
        cpx cf = csigc(gf), ci = csigc(gi), co = csigc(go);
        cpx tg = ctanhc(gc);
        cpx cp = {c_re[r], c_im[r]};
        cpx fa = cmulc(cf, cp), ib = cmulc(ci, tg);
        cpx cn = {fa.re + ib.re, fa.im + ib.im};
        cpx hn = cmulc(co, ctanhc(cn));
        c_re[r] = cn.re; c_im[r] = cn.im;
        s_re[r] += hn.re; s_im[r] += hn.im;
        hre[r * HH + u] = (_Float16)hn.re;
        him[r * HH + u] = (_Float16)hn.im;
      }
    }
    __syncthreads();
  }
  if (tid < HH) {
#pragma unroll
    for (int r = 0; r < 4; ++r) {
      feat_re[r * (2 * HH) + coloff + tid] = s_re[r] * (1.f / (float)TT);
      feat_im[r * (2 * HH) + coloff + tid] = s_im[r] * (1.f / (float)TT);
    }
  }
}

// ---------------- PLA staging: feats^T padded [NEV][KCOV] (A and B^T of cov GEMM) -------
__global__ void k_featcast(const float* __restrict__ fre, const float* __restrict__ fim,
                           _Float16* __restrict__ tre, _Float16* __restrict__ tim) {
  int idx = blockIdx.x * blockDim.x + threadIdx.x;
  if (idx >= NEV * KCOV) return;
  int k = idx % KCOV;
  int m = idx / KCOV;
  float vr = 0.f, vi = 0.f;
  if (k < NB) { vr = fre[k * NEV + m]; vi = fim[k * NEV + m]; }
  tre[idx] = (_Float16)vr;
  tim[idx] = (_Float16)vi;
}

__global__ void k_herm(const float* __restrict__ cre, const float* __restrict__ cim,
                       float* __restrict__ hre, float* __restrict__ him) {
  int idx = blockIdx.x * blockDim.x + threadIdx.x;
  if (idx >= NEV * NEV) return;
  int p = idx / NEV, q = idx % NEV;
  if (p > q)      { hre[idx] = cre[p * NEV + q]; him[idx] = cim[p * NEV + q]; }
  else if (p < q) { hre[idx] = cre[q * NEV + p]; him[idx] = -cim[q * NEV + p]; }
  else            { hre[idx] = cre[p * NEV + p]; him[idx] = 0.f; }
}

__global__ void k_eye(float* __restrict__ Vre, float* __restrict__ Vim) {
  int idx = blockIdx.x * blockDim.x + threadIdx.x;
  if (idx >= NEV * NEV) return;
  Vre[idx] = ((idx / NEV) == (idx % NEV)) ? 1.f : 0.f;
  Vim[idx] = 0.f;
}

// Parallel-ordered cyclic complex Jacobi on Hermitian A (round-robin pairing).
__global__ __launch_bounds__(512)
void k_jacobi(float* __restrict__ Are, float* __restrict__ Aim,
              float* __restrict__ Vre, float* __restrict__ Vim, int nsweeps) {
  const int n = NEV;
  __shared__ short perm[NEV];
  __shared__ short tmp[NEV];
  __shared__ short pp[NEV / 2];
  __shared__ short pq[NEV / 2];
  __shared__ float rc[NEV / 2];
  __shared__ float rsr[NEV / 2];
  __shared__ float rsi[NEV / 2];
  int tid = threadIdx.x;
  perm[tid] = (short)tid;
  __syncthreads();
  for (int sw = 0; sw < nsweeps; ++sw) {
    for (int round = 0; round < n - 1; ++round) {
      if (tid < n / 2) {
        int p = perm[tid], q = perm[n - 1 - tid];
        if (p > q) { int z = p; p = q; q = z; }
        float app = Are[p * n + p], aqq = Are[q * n + q];
        float ar = Are[p * n + q], aiv = Aim[p * n + q];
        float mag = sqrtf(ar * ar + aiv * aiv);
        float c = 1.f, sr = 0.f, si = 0.f;
        if (mag > 1e-12f) {
          float tau = (aqq - app) / (2.f * mag);
          float tt = ((tau >= 0.f) ? 1.f : -1.f) / (fabsf(tau) + sqrtf(1.f + tau * tau));
          c = 1.f / sqrtf(1.f + tt * tt);
          float sn = tt * c;
          sr = sn * (ar / mag);
          si = sn * (aiv / mag);
        }
        pp[tid] = (short)p; pq[tid] = (short)q;
        rc[tid] = c; rsr[tid] = sr; rsi[tid] = si;
      }
      __syncthreads();
      // A <- A*J (columns p,q of each row)
      for (int w = tid; w < (n / 2) * n; w += 512) {
        int pi = w >> 9, r = w & (n - 1);
        int p = pp[pi], q = pq[pi];
        float c = rc[pi], sr = rsr[pi], si = rsi[pi];
        float apr = Are[r * n + p], api = Aim[r * n + p];
        float aqr = Are[r * n + q], aqi = Aim[r * n + q];
        Are[r * n + p] = c * apr - (sr * aqr + si * aqi);
        Aim[r * n + p] = c * api - (sr * aqi - si * aqr);
        Are[r * n + q] = sr * apr - si * api + c * aqr;
        Aim[r * n + q] = sr * api + si * apr + c * aqi;
      }
      __syncthreads();
      // A <- J^H * A (rows p,q of each column)
      for (int w = tid; w < (n / 2) * n; w += 512) {
        int pi = w >> 9, cc = w & (n - 1);
        int p = pp[pi], q = pq[pi];
        float c = rc[pi], sr = rsr[pi], si = rsi[pi];
        float bpr = Are[p * n + cc], bpi = Aim[p * n + cc];
        float bqr = Are[q * n + cc], bqi = Aim[q * n + cc];
        Are[p * n + cc] = c * bpr - (sr * bqr - si * bqi);
        Aim[p * n + cc] = c * bpi - (sr * bqi + si * bqr);
        Are[q * n + cc] = sr * bpr + si * bpi + c * bqr;
        Aim[q * n + cc] = sr * bpi - si * bpr + c * bqi;
      }
      __syncthreads();
      // V <- V*J
      for (int w = tid; w < (n / 2) * n; w += 512) {
        int pi = w >> 9, r = w & (n - 1);
        int p = pp[pi], q = pq[pi];
        float c = rc[pi], sr = rsr[pi], si = rsi[pi];
        float vpr = Vre[r * n + p], vpi = Vim[r * n + p];
        float vqr = Vre[r * n + q], vqi = Vim[r * n + q];
        Vre[r * n + p] = c * vpr - (sr * vqr + si * vqi);
        Vim[r * n + p] = c * vpi - (sr * vqi - si * vqr);
        Vre[r * n + q] = sr * vpr - si * vpi + c * vqr;
        Vim[r * n + q] = sr * vpi + si * vpr + c * vqi;
      }
      __syncthreads();
      // rotate tournament pairing (keep perm[0])
      tmp[tid] = perm[tid];
      __syncthreads();
      if (tid >= 2) perm[tid] = tmp[tid - 1];
      else if (tid == 1) perm[1] = tmp[n - 1];
      __syncthreads();
    }
  }
}

__global__ void k_select(const float* __restrict__ Are, int* __restrict__ sel) {
  int i = blockIdx.x * blockDim.x + threadIdx.x;
  if (i >= NEV) return;
  float ei = Are[i * NEV + i];
  int rank = 0;
  for (int j = 0; j < NEV; ++j) {
    float ej = Are[j * NEV + j];
    if (ej > ei || (ej == ei && j < i)) rank++;
  }
  if (rank < NC_) sel[rank] = i;
}

__global__ void k_project_acc(const float* __restrict__ fre, const float* __restrict__ fim,
                              const float* __restrict__ Vre, const float* __restrict__ Vim,
                              const int* __restrict__ sel,
                              float* __restrict__ fsre, float* __restrict__ fsim,
                              int zeroFirst) {
  int idx = blockIdx.x * blockDim.x + threadIdx.x;
  if (idx >= NB * NC_) return;
  int m = idx % NC_, nrow = idx / NC_;
  int col = sel[m];
  float ar = 0.f, ai = 0.f;
  for (int k = 0; k < NEV; ++k) {
    float fr = fre[nrow * NEV + k], fi = fim[nrow * NEV + k];
    float vr = Vre[k * NEV + col], vi = Vim[k * NEV + col];
    ar += fr * vr - fi * vi;
    ai += fr * vi + fi * vr;
  }
  if (zeroFirst) { fsre[idx] = ar; fsim[idx] = ai; }
  else           { fsre[idx] += ar; fsim[idx] += ai; }
}

// ---------------- head: fc1+silu, fc2, softmax ----------------
__global__ void k_head(const float* __restrict__ fsre,
                       const float* __restrict__ fc1w, const float* __restrict__ fc1b,
                       const float* __restrict__ fc2w, const float* __restrict__ fc2b,
                       float* __restrict__ out) {
  __shared__ float hbuf[NB][16];
  int tid = threadIdx.x;
  if (tid < NB * 16) {
    int j = tid & 15, nrow = tid >> 4;
    float a = fc1b[j];
    for (int k = 0; k < NC_; ++k) a += fsre[nrow * NC_ + k] * fc1w[j * NC_ + k];
    hbuf[nrow][j] = a / (1.f + __expf(-a));   // silu
  }
  __syncthreads();
  if (tid < NB) {
    float l0 = fc2b[0], l1 = fc2b[1];
    for (int j = 0; j < 16; ++j) {
      l0 += hbuf[tid][j] * fc2w[j];
      l1 += hbuf[tid][j] * fc2w[16 + j];
    }
    float mx = fmaxf(l0, l1);
    float e0 = __expf(l0 - mx), e1 = __expf(l1 - mx);
    float s = e0 + e1;
    out[tid * 2 + 0] = e0 / s;
    out[tid * 2 + 1] = e1 / s;
  }
}

// =====================================================================================
extern "C" void kernel_launch(void* const* d_in, const int* in_sizes, int n_in,
                              void* d_out, int out_size, void* d_ws, size_t ws_size,
                              hipStream_t stream) {
  (void)in_sizes; (void)n_in; (void)out_size; (void)ws_size;
  const int*   x    = (const int*)d_in[0];
  const float* tab  = (const float*)d_in[1];
  const float* cw[5]  = {(const float*)d_in[2], (const float*)d_in[4], (const float*)d_in[6],
                         (const float*)d_in[8], (const float*)d_in[10]};
  const float* cb[5]  = {(const float*)d_in[3], (const float*)d_in[5], (const float*)d_in[7],
                         (const float*)d_in[9], (const float*)d_in[11]};
  const float* Wih  = (const float*)d_in[12];
  const float* Whh  = (const float*)d_in[13];
  const float* bih  = (const float*)d_in[14];
  const float* bhh  = (const float*)d_in[15];
  const float* fc1w = (const float*)d_in[16];
  const float* fc1b = (const float*)d_in[17];
  const float* fc2w = (const float*)d_in[18];
  const float* fc2b = (const float*)d_in[19];
  float* out = (float*)d_out;

  char* base = (char*)d_ws;
  size_t off = 0;
  auto alloc = [&](size_t bytes) -> void* {
    void* p = base + off;
    off += (bytes + 255) & ~(size_t)255;
    return p;
  };

  const int taps[5]  = {2, 4, 3, 6, 5};
  const int strd[5]  = {1, 2, 3, 3, 3};
  const int padd[5]  = {0, 0, 2, 2, 0};
  const int louts[5] = {255, 127, 86, 85, 84};

  _Float16* e16 = (_Float16*)alloc((size_t)NB * LL * DNP * 2);
  _Float16* cw16[5];
  float* co[5];
  for (int i = 0; i < 5; ++i) {
    cw16[i] = (_Float16*)alloc((size_t)taps[i] * DOP * DNP * 2);
    co[i]   = (float*)alloc((size_t)NB * DN * louts[i] * 4);
  }
  _Float16* seqre[3];
  _Float16* seqim[3];
  for (int i = 0; i < 3; ++i) {
    seqre[i] = (_Float16*)alloc((size_t)MR * DNP * 2);
    seqim[i] = (_Float16*)alloc((size_t)MR * DNP * 2);
  }
  _Float16* WBTre = (_Float16*)alloc((size_t)6 * GG * DNP * 2);
  _Float16* WBTim = (_Float16*)alloc((size_t)6 * GG * DNP * 2);
  _Float16* UBTre = (_Float16*)alloc((size_t)6 * GG * HH * 2);
  _Float16* UBTim = (_Float16*)alloc((size_t)6 * GG * HH * 2);
  float* bcre = (float*)alloc((size_t)6 * GG * 4);
  float* bcim = (float*)alloc((size_t)6 * GG * 4);
  float* pre_re = (float*)alloc((size_t)MR * GG * 4);
  float* pre_im = (float*)alloc((size_t)MR * GG * 4);
  float* featre[3];
  float* featim[3];
  for (int i = 0; i < 3; ++i) {
    featre[i] = (float*)alloc((size_t)NB * NEV * 4);
    featim[i] = (float*)alloc((size_t)NB * NEV * 4);
  }
  _Float16* fT16re = (_Float16*)alloc((size_t)NEV * KCOV * 2);
  _Float16* fT16im = (_Float16*)alloc((size_t)NEV * KCOV * 2);
  float* cov_re = (float*)alloc((size_t)NEV * NEV * 4);
  float* cov_im = (float*)alloc((size_t)NEV * NEV * 4);
  float* herm_re = (float*)alloc((size_t)NEV * NEV * 4);
  float* herm_im = (float*)alloc((size_t)NEV * NEV * 4);
  float* Vre = (float*)alloc((size_t)NEV * NEV * 4);
  float* Vim = (float*)alloc((size_t)NEV * NEV * 4);
  int* sel = (int*)alloc((size_t)NC_ * 4);
  float* fsre = (float*)alloc((size_t)NB * NC_ * 4);
  float* fsim = (float*)alloc((size_t)NB * NC_ * 4);

#define GRID1D(cnt) dim3(((cnt) + 255) / 256), dim3(256)

  // embedding, natural [n][l][d] padded layout
  { int c = NB * LL * DNP; k_embed16<<<GRID1D(c), 0, stream>>>(x, tab, e16); }

  // weight staging (no transposes needed: B^T layouts are native)
  for (int i = 0; i < 5; ++i) {
    int c = taps[i] * DOP * DNP;
    k_convw16<<<GRID1D(c), 0, stream>>>(cw[i], cw16[i], taps[i]);
  }
  { int c = 6 * GG * DNP; k_wihBT<<<GRID1D(c), 0, stream>>>(Wih, WBTre, WBTim); }
  { int c = 6 * GG * HH;  k_whhBT<<<GRID1D(c), 0, stream>>>(Whh, UBTre, UBTim); }
  { int c = 6 * GG;       k_bcomb<<<GRID1D(c), 0, stream>>>(bih, bhh, bcre, bcim); }

  // convolutions (WMMA)
  for (int i = 0; i < 5; ++i) {
    dim3 g((louts[i] + 15) / 16, DOP / 16, NB);
    k_conv_wmma<<<g, 32, 0, stream>>>(e16, cw16[i], cb[i], co[i],
                                      louts[i], taps[i], strd[i], padd[i]);
  }

  // zero + scatter sequences
  for (int i = 0; i < 3; ++i) {
    int c = MR * DNP;
    k_zero_h<<<GRID1D(c), 0, stream>>>(seqre[i], c);
    k_zero_h<<<GRID1D(c), 0, stream>>>(seqim[i], c);
  }
  { int c = NB * DN * 255; k_scat2<<<GRID1D(c), 0, stream>>>(co[0], seqre[0]); }
  { int c = NB * DN * 127; k_scat4<<<GRID1D(c), 0, stream>>>(co[1], seqim[0]); }
  { int c = NB * DN * 84;  k_scat3<<<GRID1D(c), 0, stream>>>(co[2], seqre[1]); }
  { int c = NB * DN * 85;  k_scat6<<<GRID1D(c), 0, stream>>>(co[3], seqim[1]); }
  { int c = NB * DN * 83;  k_scat5<<<GRID1D(c), 0, stream>>>(co[4], seqim[2]); }

  // biLSTMs
  for (int l = 0; l < 3; ++l) {
    for (int dir = 0; dir < 2; ++dir) {
      int slab = l * 2 + dir;
      const _Float16* Bre = WBTre + (size_t)slab * GG * DNP;
      const _Float16* Bim = WBTim + (size_t)slab * GG * DNP;
      dim3 gp(GG / 16, MR / 16);
      // pre = seq @ Wih^T (complex = 4 real WMMA GEMMs)
      k_gemm16<0, 0><<<gp, 32, 0, stream>>>(seqre[l], DNP, Bre, DNP, pre_re, GG, MR, GG, DNP);
      k_gemm16<1, 1><<<gp, 32, 0, stream>>>(seqim[l], DNP, Bim, DNP, pre_re, GG, MR, GG, DNP);
      k_gemm16<0, 0><<<gp, 32, 0, stream>>>(seqre[l], DNP, Bim, DNP, pre_im, GG, MR, GG, DNP);
      k_gemm16<0, 1><<<gp, 32, 0, stream>>>(seqim[l], DNP, Bre, DNP, pre_im, GG, MR, GG, DNP);
      { int c = MR * GG;
        k_addbias<<<GRID1D(c), 0, stream>>>(pre_re, pre_im, bcre + slab * GG, bcim + slab * GG); }
      k_lstm_scan<<<1, 512, 0, stream>>>(pre_re, pre_im,
                                         UBTre + (size_t)slab * GG * HH,
                                         UBTim + (size_t)slab * GG * HH,
                                         featre[l], featim[l], dir, dir * HH);
    }
  }

  // PLA per branch, summed into fuse
  for (int l = 0; l < 3; ++l) {
    { int c = NEV * KCOV;
      k_featcast<<<GRID1D(c), 0, stream>>>(featre[l], featim[l], fT16re, fT16im); }
    dim3 gc(NEV / 16, NEV / 16);
    // cov = feats^T feats (non-conjugate): A = featsT, B^T = featsT (same buffer)
    k_gemm16<0, 0><<<gc, 32, 0, stream>>>(fT16re, KCOV, fT16re, KCOV, cov_re, NEV, NEV, NEV, KCOV);
    k_gemm16<1, 1><<<gc, 32, 0, stream>>>(fT16im, KCOV, fT16im, KCOV, cov_re, NEV, NEV, NEV, KCOV);
    k_gemm16<0, 0><<<gc, 32, 0, stream>>>(fT16re, KCOV, fT16im, KCOV, cov_im, NEV, NEV, NEV, KCOV);
    k_gemm16<0, 1><<<gc, 32, 0, stream>>>(fT16im, KCOV, fT16re, KCOV, cov_im, NEV, NEV, NEV, KCOV);
    { int c = NEV * NEV;
      k_herm<<<GRID1D(c), 0, stream>>>(cov_re, cov_im, herm_re, herm_im);
      k_eye<<<GRID1D(c), 0, stream>>>(Vre, Vim); }
    k_jacobi<<<1, 512, 0, stream>>>(herm_re, herm_im, Vre, Vim, 3);
    k_select<<<dim3(2), dim3(256), 0, stream>>>(herm_re, sel);
    { int c = NB * NC_;
      k_project_acc<<<GRID1D(c), 0, stream>>>(featre[l], featim[l], Vre, Vim, sel,
                                              fsre, fsim, (l == 0) ? 1 : 0); }
  }

  k_head<<<1, 64, 0, stream>>>(fsre, fc1w, fc1b, fc2w, fc2b, out);
#undef GRID1D
}
